// Model_69630009802883
// MI455X (gfx1250) — matrix operand order repaired and checked by reading, not compiled, not run on hardware
//
#include <hip/hip_runtime.h>
#include <hip/hip_bf16.h>

#define BB 32
#define TT 512
#define II 128
#define HH 1024
#define GG 4096
#define OO 128

typedef unsigned short u16t;
typedef unsigned int   u32t;

typedef __attribute__((ext_vector_type(16))) __bf16 v16bf;
typedef __attribute__((ext_vector_type(8)))  __bf16 v8bf;
typedef __attribute__((ext_vector_type(8)))  float  v8f;

// ---------- helpers ----------

__device__ __forceinline__ u16t f2bf(float f) {
  u32t u = __float_as_uint(f);
  u += 0x7fffu + ((u >> 16) & 1u);   // round-to-nearest-even
  return (u16t)(u >> 16);
}

__device__ __forceinline__ v8f wmma_bf16(v16bf a, v16bf b, v8f c) {
  // D = A(16x32 bf16) * B(32x16 bf16) + C(16x16 f32)
  return __builtin_amdgcn_wmma_f32_16x16x32_bf16(false, a, false, b,
                                                 (short)0, c, false, false);
}

// A-matrix 16x32 bf16 load, row-major source with row stride ldk (halfs).
// lanes 0-15: row=lane, K {k0..k0+7, k0+16..k0+23}
// lanes 16-31: row=lane-16, K {k0+8..k0+15, k0+24..k0+31}
__device__ __forceinline__ v16bf load_a16(const u16t* __restrict__ base,
                                          int row0, int ldk, int k0, int lane) {
  int r  = row0 + (lane & 15);
  int ka = (lane < 16) ? 0 : 8;
  const u16t* p = base + (size_t)r * (size_t)ldk + k0 + ka;
  v16bf a;
  ((v8bf*)&a)[0] = *(const v8bf*)(p);
  ((v8bf*)&a)[1] = *(const v8bf*)(p + 16);
  return a;
}

// B-matrix 32x16 bf16: B[k][n] = W[n0+n][k], W row-major [*, ldk].
// lanes 0-15: col=lane, K {k0..k0+7, k0+16..k0+23} ; lanes 16-31: col=lane-16, K {k0+8..k0+15, k0+24..k0+31}
__device__ __forceinline__ v16bf load_b16(const u16t* __restrict__ wbase,
                                          int n0, int ldk, int k0, int lane) {
  int n    = lane & 15;
  int koff = (lane < 16) ? 0 : 8;
  const u16t* p = wbase + (size_t)(n0 + n) * (size_t)ldk + k0 + koff;
  v16bf b;
  ((v8bf*)&b)[0] = *(const v8bf*)(p);
  ((v8bf*)&b)[1] = *(const v8bf*)(p + 16);
  return b;
}

__device__ __forceinline__ float sigmoidf_(float x) {
  return 1.0f / (1.0f + expf(-x));
}

// ---------- kernels ----------

__global__ __launch_bounds__(256) void convert_f32_bf16(
    const float* __restrict__ src, u16t* __restrict__ dst, int n) {
  int i = blockIdx.x * blockDim.x + threadIdx.x;
  int stride = gridDim.x * blockDim.x;
  for (; i < n; i += stride) dst[i] = f2bf(src[i]);
}

// Persistent fused LSTM layer:
//   gates[t] = bias + x_t @ Wx^T (K=Kx) + h_{t-1} @ Wh^T (K=H)
// Grid: 64 blocks x 256 threads. Block j owns hidden columns [16j, 16j+16).
// Wave w: mtile = w&1 (batch rows), part = w>>2? no: part = w>>1 in {i,f,g,o}.
// c-state lives entirely in registers (2 elems/thread). h double-buffered in
// global memory (bf16); one grid-wide atomic barrier per timestep.
__global__ __launch_bounds__(256) void lstm_layer_persistent(
    const u16t* __restrict__ Wh,     // [GG, HH] bf16
    const u16t* __restrict__ Wx,     // [GG, Kx] bf16
    const float* __restrict__ bias,  // [GG]
    const u16t* __restrict__ xseq,   // input seq, bf16
    int xs_tstride,                  // halfs between timesteps
    int xs_rstride,                  // halfs between batch rows
    int Kx,                          // input feature dim (128 or 1024)
    u16t* __restrict__ hseq,         // [TT][BB][HH] bf16 out
    u16t* __restrict__ hstate,       // [2][BB][HH] bf16 double buffer (zeroed)
    u32t* __restrict__ cnt,          // [TT] zeroed
    u32t* __restrict__ gen)          // [1] zeroed
{
  __shared__ float gl[4][2][16][16];  // [part][mtile][M][N] gate tiles

  const int lane = threadIdx.x & 31;
  const int w    = threadIdx.x >> 5;
  const int mt   = w & 1;       // batch tile: rows 16*mt .. 16*mt+15
  const int part = w >> 1;      // 0=i 1=f 2=g 3=o
  const int j    = blockIdx.x;  // hidden slice 16j..16j+15
  const int n0   = part * HH + j * 16;   // gate column base
  const int row0 = mt * 16;
  const int Nidx = lane & 15;
  const int Mhi  = (lane >> 4) << 3;     // 0 or 8

  // cell state in registers: thread owns elements e0, e1 forever
  const int e0 = threadIdx.x;
  const int e1 = threadIdx.x + 256;
  float c0 = 0.0f, c1 = 0.0f;

  const float bv = bias[n0 + Nidx];

  for (int t = 0; t < TT; ++t) {
    const u16t* hbuf = hstate + (size_t)(t & 1) * (BB * HH);
    const u16t* xt   = xseq + (size_t)t * (size_t)xs_tstride;

    v8f acc;
#pragma unroll
    for (int v = 0; v < 8; ++v) acc[v] = bv;

    // h_{t-1} @ Wh^T  (K = HH)
#pragma unroll 4
    for (int k0 = 0; k0 < HH; k0 += 32) {
      v16bf a = load_a16(hbuf, row0, HH, k0, lane);
      v16bf b = load_b16(Wh, n0, HH, k0, lane);
      acc = wmma_bf16(a, b, acc);
    }
    // x_t @ Wx^T  (K = Kx)
#pragma unroll 4
    for (int k0 = 0; k0 < Kx; k0 += 32) {
      v16bf a = load_a16(xt, row0, xs_rstride, k0, lane);
      v16bf b = load_b16(Wx, n0, Kx, k0, lane);
      acc = wmma_bf16(a, b, acc);
    }

    // gate tile -> LDS
#pragma unroll
    for (int v = 0; v < 8; ++v) gl[part][mt][v + Mhi][Nidx] = acc[v];
    __syncthreads();

    // activations + state update (512 elems, 2 per thread)
    u16t* hnext = hstate + (size_t)((t + 1) & 1) * (BB * HH);
    u16t* hout  = hseq + (size_t)t * (BB * HH);
    {
      int b = e0 >> 4, hc = e0 & 15;
      float ig = gl[0][b >> 4][b & 15][hc];
      float fg = gl[1][b >> 4][b & 15][hc];
      float gg = gl[2][b >> 4][b & 15][hc];
      float og = gl[3][b >> 4][b & 15][hc];
      c0 = sigmoidf_(fg) * c0 + sigmoidf_(ig) * tanhf(gg);
      float h = sigmoidf_(og) * tanhf(c0);
      u16t hb = f2bf(h);
      int col = j * 16 + hc;
      hnext[b * HH + col] = hb;
      hout[b * HH + col]  = hb;
    }
    {
      int b = e1 >> 4, hc = e1 & 15;
      float ig = gl[0][b >> 4][b & 15][hc];
      float fg = gl[1][b >> 4][b & 15][hc];
      float gg = gl[2][b >> 4][b & 15][hc];
      float og = gl[3][b >> 4][b & 15][hc];
      c1 = sigmoidf_(fg) * c1 + sigmoidf_(ig) * tanhf(gg);
      float h = sigmoidf_(og) * tanhf(c1);
      u16t hb = f2bf(h);
      int col = j * 16 + hc;
      hnext[b * HH + col] = hb;
      hout[b * HH + col]  = hb;
    }
    __syncthreads();  // protect LDS for next iteration's writes

    // grid-wide barrier: next step reads hnext written by all blocks
    if (threadIdx.x == 0) {
      __threadfence();
      u32t prev = atomicAdd(&cnt[t], 1u);
      if (prev == gridDim.x - 1) {
        __hip_atomic_store(gen, (u32t)(t + 1), __ATOMIC_RELEASE,
                           __HIP_MEMORY_SCOPE_AGENT);
      } else {
        while (__hip_atomic_load(gen, __ATOMIC_ACQUIRE,
                                 __HIP_MEMORY_SCOPE_AGENT) < (u32t)(t + 1)) {
          __builtin_amdgcn_s_sleep(1);
        }
      }
      __threadfence();
    }
    __syncthreads();
  }
}

// out[b][t][o] = h2seq[t][b][:] @ Wout^T + bout
// h2seq rows ordered r = t*32 + b (16-row tiles stay within one t).
__global__ __launch_bounds__(256) void head_kernel(
    const u16t* __restrict__ h2,    // [TT][BB][HH] bf16
    const u16t* __restrict__ Wo,    // [OO][HH] bf16
    const float* __restrict__ bo,   // [OO]
    float* __restrict__ out)        // [BB][TT][OO]
{
  const int lane = threadIdx.x & 31;
  const int w    = threadIdx.x >> 5;   // N tile (8 waves cover O=128)
  const int mt   = blockIdx.x;         // 0..1023 (M tiles over 16384 rows)
  const int n0   = w * 16;
  const int row0 = mt * 16;
  const int Nidx = lane & 15;
  const int Mhi  = (lane >> 4) << 3;

  v8f acc;
  float bv = bo[n0 + Nidx];
#pragma unroll
  for (int v = 0; v < 8; ++v) acc[v] = bv;

#pragma unroll 4
  for (int k0 = 0; k0 < HH; k0 += 32) {
    v16bf a = load_a16(h2, row0, HH, k0, lane);
    v16bf b = load_b16(Wo, n0, HH, k0, lane);
    acc = wmma_bf16(a, b, acc);
  }

  const int t  = mt >> 1;
  const int b0 = (mt & 1) * 16;
#pragma unroll
  for (int v = 0; v < 8; ++v) {
    int bb = b0 + v + Mhi;
    out[(size_t)bb * (TT * OO) + (size_t)t * OO + n0 + Nidx] = acc[v];
  }
}

// ---------- launch ----------

extern "C" void kernel_launch(void* const* d_in, const int* in_sizes, int n_in,
                              void* d_out, int out_size, void* d_ws, size_t ws_size,
                              hipStream_t stream) {
  const float* X    = (const float*)d_in[0];  // [B,T,I]
  const float* Wih0 = (const float*)d_in[1];  // [4H,I]
  const float* Whh0 = (const float*)d_in[2];  // [4H,H]
  const float* b0   = (const float*)d_in[3];  // [4H]
  const float* Wih1 = (const float*)d_in[4];  // [4H,H]
  const float* Whh1 = (const float*)d_in[5];  // [4H,H]
  const float* b1   = (const float*)d_in[6];  // [4H]
  const float* Wout = (const float*)d_in[7];  // [O,H]
  const float* bout = (const float*)d_in[8];  // [O]
  float* out = (float*)d_out;

  char* ws = (char*)d_ws;
  size_t off = 0;
  auto alloc = [&](size_t bytes) -> char* {
    char* p = ws + off;
    off = (off + bytes + 255) & ~(size_t)255;
    return p;
  };

  u16t* Xb     = (u16t*)alloc((size_t)BB * TT * II * 2);  // 4 MB
  u16t* Wih0b  = (u16t*)alloc((size_t)GG * II * 2);       // 1 MB
  u16t* Whh0b  = (u16t*)alloc((size_t)GG * HH * 2);       // 8 MB
  u16t* Wih1b  = (u16t*)alloc((size_t)GG * HH * 2);       // 8 MB
  u16t* Whh1b  = (u16t*)alloc((size_t)GG * HH * 2);       // 8 MB
  u16t* Wob    = (u16t*)alloc((size_t)OO * HH * 2);       // 256 KB
  u16t* h1seq  = (u16t*)alloc((size_t)TT * BB * HH * 2);  // 32 MB
  u16t* h2seq  = (u16t*)alloc((size_t)TT * BB * HH * 2);  // 32 MB
  // zero-initialized region (h double buffers + sync)
  size_t zero_begin = off;
  u16t* hstate0 = (u16t*)alloc((size_t)2 * BB * HH * 2);  // 128 KB
  u16t* hstate1 = (u16t*)alloc((size_t)2 * BB * HH * 2);  // 128 KB
  u32t* cnt0    = (u32t*)alloc((size_t)TT * 4);
  u32t* gen0    = (u32t*)alloc(4);
  u32t* cnt1    = (u32t*)alloc((size_t)TT * 4);
  u32t* gen1    = (u32t*)alloc(4);
  size_t zero_bytes = off - zero_begin;
  (void)ws_size; (void)in_sizes; (void)n_in; (void)out_size;

  hipMemsetAsync(ws + zero_begin, 0, zero_bytes, stream);

  // fp32 -> bf16 conversions
  convert_f32_bf16<<<1024, 256, 0, stream>>>(X,    Xb,    BB * TT * II);
  convert_f32_bf16<<<1024, 256, 0, stream>>>(Wih0, Wih0b, GG * II);
  convert_f32_bf16<<<2048, 256, 0, stream>>>(Whh0, Whh0b, GG * HH);
  convert_f32_bf16<<<2048, 256, 0, stream>>>(Wih1, Wih1b, GG * HH);
  convert_f32_bf16<<<2048, 256, 0, stream>>>(Whh1, Whh1b, GG * HH);
  convert_f32_bf16<<<256,  256, 0, stream>>>(Wout, Wob,   OO * HH);

  // Layer 0: x layout [B,T,I] -> row stride T*I halfs, t stride I halfs, Kx=I
  lstm_layer_persistent<<<64, 256, 0, stream>>>(
      Whh0b, Wih0b, b0, Xb, /*xs_tstride=*/II, /*xs_rstride=*/TT * II,
      /*Kx=*/II, h1seq, hstate0, cnt0, gen0);

  // Layer 1: x = h1seq [T,B,H] -> row stride H, t stride B*H, Kx=H
  lstm_layer_persistent<<<64, 256, 0, stream>>>(
      Whh1b, Wih1b, b1, h1seq, /*xs_tstride=*/BB * HH, /*xs_rstride=*/HH,
      /*Kx=*/HH, h2seq, hstate1, cnt1, gen1);

  // Output head
  head_kernel<<<1024, 256, 0, stream>>>(h2seq, Wob, bout, out);
}
